// ModernBertAttention_87411174408281
// MI455X (gfx1250) — matrix operand order repaired and checked by reading, not compiled, not run on hardware
//
#include <hip/hip_runtime.h>

// ---------------- CDNA5 WMMA types ----------------
typedef __attribute__((ext_vector_type(16))) __bf16 v16bf;
typedef __attribute__((ext_vector_type(8)))  float  v8f;

__device__ __forceinline__ v8f wmma_bf16(v16bf a, v16bf b, v8f c) {
    // (neg_a, A, neg_b, B, c_mod, C, reuse_a, reuse_b)
    return __builtin_amdgcn_wmma_f32_16x16x32_bf16(false, a, false, b, (short)0, c, false, false);
}

// ---------------- CDNA5 async global->LDS copy (ASYNCcnt path) ----------------
// GLOBAL_LOAD_ASYNC_TO_LDS_B128 (ISA 15.18.3 opcode 98): per-lane copy of 16B
// from global memory directly into LDS; tracked by ASYNCcnt.
__device__ __forceinline__ void async_load_b128(unsigned lds_addr, const void* gaddr) {
    unsigned long long ga = (unsigned long long)(uintptr_t)gaddr;
    asm volatile("global_load_async_to_lds_b128 %0, %1, off"
                 :: "v"(lds_addr), "v"(ga) : "memory");
}
__device__ __forceinline__ void wait_asynccnt0() {
    asm volatile("s_wait_asynccnt 0x0" ::: "memory");
}
__device__ __forceinline__ unsigned lds_addr_of(const void* p) {
    // generic pointer to __shared__: addr[31:0] is the LDS byte offset
    return (unsigned)(uintptr_t)p;
}

#define BATCH   2
#define SEQ     2048
#define HID     1024
#define NHEADS  16
#define HD      64
#define WINDOW  128

// ---------------- fp32 -> bf16 transpose-convert ----------------
// src: [R][C] fp32 row-major  ->  dst: [C][R] bf16 (K-contiguous weight layout)
__global__ void cvt_transpose_kernel(const float* __restrict__ src,
                                     __bf16* __restrict__ dst, int R, int C) {
    int i = blockIdx.x * blockDim.x + threadIdx.x;
    if (i >= R * C) return;
    int r = i / C, c = i % C;                  // coalesced read
    dst[(size_t)c * R + r] = (__bf16)src[i];   // strided one-time write
}

// ---------------- RoPE + convert ----------------
__global__ void rope_convert_kernel(const float* __restrict__ x,
                                    __bf16* __restrict__ qin,
                                    __bf16* __restrict__ xbf,
                                    int total) {
    int i = blockIdx.x * blockDim.x + threadIdx.x;
    if (i >= total) return;
    float v = x[i];
    xbf[i] = (__bf16)v;
    int d  = i & (HID - 1);          // position within hidden
    int s  = (i >> 10) & (SEQ - 1);  // sequence position
    int j  = d & (HD - 1);           // dim within head
    int io = j & 31;                 // freq index
    float inv_freq = __powf(10000.0f, -(float)(2 * io) / 64.0f);
    float ang = (float)s * inv_freq;
    float sn, cs;
    __sincosf(ang, &sn, &cs);
    float other = (j < 32) ? -x[i + 32] : x[i - 32];
    qin[i] = (__bf16)(v * cs + other * sn);
}

// ---------------- generic bf16 WMMA GEMM (async double-buffered) ----------------
// C[M,N] = A[M,K] * W[N,K]^T.  A row-major bf16, W TRANSPOSED (K-contiguous).
// mode 0: f32 row-major | mode 1: bf16 [B,H,S,HD] (Q,K) | mode 2: bf16 [B,H,HD,S] (V)
__global__ void __launch_bounds__(128)
gemm_bf16_kernel(const __bf16* __restrict__ A, const __bf16* __restrict__ WT,
                 void* __restrict__ Cout, int M, int N, int K,
                 int mode, float scale) {
    __shared__ __bf16 As[2 * 2048];   // double-buffered 64x32 [m][k]
    __shared__ __bf16 Bs[2 * 2048];   // double-buffered 64x32 [n][k]

    const int t    = threadIdx.x;
    const int lane = t & 31;
    const int wave = t >> 5;
    const int m0   = blockIdx.x * 64;
    const int n0   = blockIdx.y * 64;

    const int m    = lane & 15;
    const int hf   = lane >> 4;   // half-wave select
    const int kb   = hf * 8;      // unused; A fragment also uses krow
    const int krow = hf * 16;     // B fragment K-base

    // per-thread cooperative-copy coordinates (both tiles are 64x32)
    const int li = t * 16;        // element index; 32 bytes per thread
    const int lr = li >> 5;       // row within tile
    const int lc = li & 31;       // col within tile
    const __bf16* ga = A  + (size_t)(m0 + lr) * K + lc;
    const __bf16* gb = WT + (size_t)(n0 + lr) * K + lc;
    const unsigned asb = lds_addr_of(As) + li * 2;   // byte addr in buffer 0
    const unsigned bsb = lds_addr_of(Bs) + li * 2;

    v8f acc[4];
#pragma unroll
    for (int j = 0; j < 4; ++j) acc[j] = (v8f){};

    // preload tile 0 into buffer 0 via async DMA
    async_load_b128(asb,      ga);
    async_load_b128(asb + 16, ga + 8);
    async_load_b128(bsb,      gb);
    async_load_b128(bsb + 16, gb + 8);
    wait_asynccnt0();
    __syncthreads();

    const int nsteps = K / 32;
    int p = 0;
    for (int step = 0; step < nsteps; ++step) {
        const int k0 = step * 32;
        const bool has_next = (step + 1 < nsteps);
        if (has_next) {
            const unsigned dst = (unsigned)((p ^ 1) * 4096);  // buffer byte offset
            const __bf16* ga2 = ga + k0 + 32;
            const __bf16* gb2 = gb + k0 + 32;
            async_load_b128(asb + dst,      ga2);
            async_load_b128(asb + dst + 16, ga2 + 8);
            async_load_b128(bsb + dst,      gb2);
            async_load_b128(bsb + dst + 16, gb2 + 8);
            __builtin_prefetch(ga2 + 32, 0, 0);   // warm L2 for step+2
            __builtin_prefetch(gb2 + 32, 0, 0);
        }

        // ---- compute on buffer p: batch all fragment loads, then 4 WMMAs ----
        const __bf16* Abuf = As + p * 2048;
        const __bf16* Bbuf = Bs + p * 2048;

        v16bf a;
        {
            const __bf16* arow = Abuf + (wave * 16 + m) * 32;
#pragma unroll
            for (int e = 0; e < 8; ++e) {
                a[e]     = arow[krow + e];
                a[8 + e] = arow[krow + 8 + e];
            }
        }
        v16bf bfrag[4];
#pragma unroll
        for (int j = 0; j < 4; ++j) {
            const __bf16* brow = Bbuf + (j * 16 + m) * 32 + krow;
#pragma unroll
            for (int e = 0; e < 16; ++e) bfrag[j][e] = brow[e];
        }
#pragma unroll
        for (int j = 0; j < 4; ++j) acc[j] = wmma_bf16(a, bfrag[j], acc[j]);

        if (has_next) wait_asynccnt0();
        __syncthreads();
        p ^= 1;
    }

    // store: C lane layout -> row = r + 8*hf (+ wave*16), col = j*16 + (lane&15)
    if (mode == 0) {
        float* Out = (float*)Cout;
#pragma unroll
        for (int j = 0; j < 4; ++j) {
#pragma unroll
            for (int r = 0; r < 8; ++r) {
                int row = m0 + wave * 16 + r + 8 * hf;
                int col = n0 + j * 16 + m;
                Out[(size_t)row * N + col] = acc[j][r] * scale;
            }
        }
    } else if (mode == 1) {
        __bf16* Out = (__bf16*)Cout;
#pragma unroll
        for (int j = 0; j < 4; ++j) {
#pragma unroll
            for (int r = 0; r < 8; ++r) {
                int row = m0 + wave * 16 + r + 8 * hf;   // = b*SEQ + s
                int col = n0 + j * 16 + m;               // = h*HD + d
                int b_ = row >> 11;
                int s_ = row & (SEQ - 1);
                int h_ = col >> 6;
                int d_ = col & (HD - 1);
                size_t idx = (((size_t)b_ * NHEADS + h_) * SEQ + s_) * HD + d_;
                Out[idx] = (__bf16)(acc[j][r] * scale);
            }
        }
    } else {  // mode 2: V transposed per-head [B,H,HD,S]
        __bf16* Out = (__bf16*)Cout;
#pragma unroll
        for (int j = 0; j < 4; ++j) {
#pragma unroll
            for (int r = 0; r < 8; ++r) {
                int row = m0 + wave * 16 + r + 8 * hf;
                int col = n0 + j * 16 + m;
                int b_ = row >> 11;
                int s_ = row & (SEQ - 1);
                int h_ = col >> 6;
                int d_ = col & (HD - 1);
                size_t idx = (((size_t)b_ * NHEADS + h_) * HD + d_) * SEQ + s_;
                Out[idx] = (__bf16)(acc[j][r] * scale);
            }
        }
    }
}

// ---------------- sliding-window flash attention ----------------
// Q/K in [BATCH, NHEADS, SEQ, HD]; V in [BATCH, NHEADS, HD, SEQ] (transposed).
// Q pre-scaled by 1/sqrt(HD). 32 keys per iteration.
__global__ void __launch_bounds__(32)
attn_window_kernel(const __bf16* __restrict__ Qb, const __bf16* __restrict__ Kb,
                   const __bf16* __restrict__ Vt, __bf16* __restrict__ Ob) {
    __shared__ __bf16 Ptile[16 * 32];   // [row][key-in-pair]

    const int lane = threadIdx.x & 31;
    const int nqt  = SEQ / 16;
    const int bh   = blockIdx.x / nqt;   // b*NHEADS + h
    const int qt   = blockIdx.x % nqt;
    const int q0   = qt * 16;

    const __bf16* Qh = Qb + (size_t)bh * SEQ * HD;
    const __bf16* Kh = Kb + (size_t)bh * SEQ * HD;
    const __bf16* Vh = Vt + (size_t)bh * HD * SEQ;

    const int m    = lane & 15;
    const int hf   = lane >> 4;
    const int kb   = hf * 8;
    const int koff = hf * 16;

    // Q A-fragments for K-dims 0..31 and 32..63
    v16bf aq0, aq1;
    {
        const __bf16* qrow = Qh + (size_t)(q0 + m) * HD;
#pragma unroll
        for (int e = 0; e < 8; ++e) {
            aq0[e]     = qrow[koff + e];
            aq0[8 + e] = qrow[koff + 8 + e];
            aq1[e]     = qrow[32 + koff + e];
            aq1[8 + e] = qrow[40 + koff + e];
        }
    }

    float rmax[8], rsum[8];
    v8f O[4];
#pragma unroll
    for (int r = 0; r < 8; ++r) { rmax[r] = -3.0e38f; rsum[r] = 0.0f; }
#pragma unroll
    for (int c = 0; c < 4; ++c) O[c] = (v8f){};

    int klo = q0 - (WINDOW - 1); if (klo < 0) klo = 0;
    int khi = q0 + 15 + (WINDOW - 1); if (khi > SEQ - 1) khi = SEQ - 1;
    const int kstart = (klo / 32) * 32;   // pair-aligned; extra keys get masked

    for (int k0 = kstart; k0 <= khi; k0 += 32) {
        v16bf bkA0, bkA1, bkB0, bkB1;
        {
            const __bf16* krA = Kh + (size_t)(k0 + m) * HD + koff;
            const __bf16* krB = Kh + (size_t)(k0 + 16 + m) * HD + koff;
#pragma unroll
            for (int e = 0; e < 16; ++e) {
                bkA0[e] = krA[e];
                bkA1[e] = krA[32 + e];
                bkB0[e] = krB[e];
                bkB1[e] = krB[32 + e];
            }
        }
        v8f scA = (v8f){}, scB = (v8f){};
        scA = wmma_bf16(aq0, bkA0, scA);
        scA = wmma_bf16(aq1, bkA1, scA);
        scB = wmma_bf16(aq0, bkB0, scB);
        scB = wmma_bf16(aq1, bkB1, scB);

        // mask + online softmax over the 32-key pair
#pragma unroll
        for (int r = 0; r < 8; ++r) {
            int qi = q0 + r + 8 * hf;
            int kA = k0 + m;
            int kB = k0 + 16 + m;
            float sA = scA[r];
            float sB = scB[r];
            int dA = qi - kA; if (dA < 0) dA = -dA;
            int dB = qi - kB; if (dB < 0) dB = -dB;
            if (dA >= WINDOW) sA = -1.0e9f;
            if (dB >= WINDOW) sB = -1.0e9f;

            float mx = fmaxf(sA, sB);
#pragma unroll
            for (int off = 1; off < 16; off <<= 1)
                mx = fmaxf(mx, __shfl_xor(mx, off, 32));
            float Mnew  = fmaxf(rmax[r], mx);
            float alpha = __expf(rmax[r] - Mnew);
            float pA    = __expf(sA - Mnew);
            float pB    = __expf(sB - Mnew);
            float ps = pA + pB;
#pragma unroll
            for (int off = 1; off < 16; off <<= 1)
                ps += __shfl_xor(ps, off, 32);
            rsum[r] = rsum[r] * alpha + ps;
            rmax[r] = Mnew;
#pragma unroll
            for (int c = 0; c < 4; ++c) O[c][r] *= alpha;

            int prow = (r + 8 * hf) * 32;
            Ptile[prow + m]      = (__bf16)pA;
            Ptile[prow + 16 + m] = (__bf16)pB;
        }
        __syncthreads();

        // P A-fragment: full 16x32
        v16bf ap;
        {
            const __bf16* prow = Ptile + m * 32;
#pragma unroll
            for (int e = 0; e < 8; ++e) {
                ap[e]     = prow[koff + e];
                ap[8 + e] = prow[koff + 8 + e];
            }
        }

        // V B-fragments from transposed V: col n = dim, 16 contiguous keys/lane
#pragma unroll
        for (int c = 0; c < 4; ++c) {
            const __bf16* vrow = Vh + (size_t)(c * 16 + m) * SEQ + k0 + koff;
            v16bf bv;
#pragma unroll
            for (int e = 0; e < 16; ++e) bv[e] = vrow[e];
            O[c] = wmma_bf16(ap, bv, O[c]);
        }
        __syncthreads();
    }

    // normalize + store to [BATCH, SEQ, NHEADS*HD]
    const int b_ = bh >> 4;
    const int h_ = bh & 15;
#pragma unroll
    for (int r = 0; r < 8; ++r) {
        float inv = 1.0f / rsum[r];
        int s_ = q0 + r + 8 * hf;
#pragma unroll
        for (int c = 0; c < 4; ++c) {
            int d_ = c * 16 + m;
            Ob[((size_t)b_ * SEQ + s_) * (NHEADS * HD) + h_ * HD + d_] =
                (__bf16)(O[c][r] * inv);
        }
    }
}

// ---------------- launch ----------------
extern "C" void kernel_launch(void* const* d_in, const int* in_sizes, int n_in,
                              void* d_out, int out_size, void* d_ws, size_t ws_size,
                              hipStream_t stream) {
    const float* x_f32  = (const float*)d_in[0];  // [2,2048,1024]
    const float* Wq_f32 = (const float*)d_in[1];  // [K=1024][N=1024]
    const float* Wk_f32 = (const float*)d_in[2];
    const float* Wv_f32 = (const float*)d_in[3];
    const float* Wo_f32 = (const float*)d_in[4];  // [K=1024][N=1024]
    float* out = (float*)d_out;

    const size_t actElems = (size_t)BATCH * SEQ * HID;   // 4,194,304
    const size_t wElems   = (size_t)HID * HID;           // 1,048,576
    const size_t MiB      = 1024ull * 1024ull;

    char* ws = (char*)d_ws;
    __bf16* qin_bf = (__bf16*)(ws + 0);          // 8 MiB, reused as attn output
    __bf16* x_bf   = (__bf16*)(ws + 8  * MiB);   // 8 MiB
    __bf16* WqT    = (__bf16*)(ws + 16 * MiB);   // 2 MiB each, transposed [N][K]
    __bf16* WkT    = (__bf16*)(ws + 18 * MiB);
    __bf16* WvT    = (__bf16*)(ws + 20 * MiB);
    __bf16* WoT    = (__bf16*)(ws + 22 * MiB);
    __bf16* Qbuf   = (__bf16*)(ws + 24 * MiB);   // [B,H,S,64]
    __bf16* Kbuf   = (__bf16*)(ws + 32 * MiB);   // [B,H,S,64]
    __bf16* Vbuf   = (__bf16*)(ws + 40 * MiB);   // [B,H,64,S] (transposed)
    __bf16* attn   = qin_bf;                     // reuse after QK projections

    // 1) weight conversion + transpose to [N][K] bf16
    {
        dim3 blk(256), grd((unsigned)((wElems + 255) / 256));
        cvt_transpose_kernel<<<grd, blk, 0, stream>>>(Wq_f32, WqT, HID, HID);
        cvt_transpose_kernel<<<grd, blk, 0, stream>>>(Wk_f32, WkT, HID, HID);
        cvt_transpose_kernel<<<grd, blk, 0, stream>>>(Wv_f32, WvT, HID, HID);
        cvt_transpose_kernel<<<grd, blk, 0, stream>>>(Wo_f32, WoT, HID, HID);
    }

    // 2) RoPE + convert
    {
        dim3 blk(256), grd((unsigned)((actElems + 255) / 256));
        rope_convert_kernel<<<grd, blk, 0, stream>>>(x_f32, qin_bf, x_bf, (int)actElems);
    }

    // 3) QKV projections: [4096,1024] x [1024,1024]
    {
        const int M = BATCH * SEQ, N = HID, K = HID;
        dim3 blk(128), grd(M / 64, N / 64);
        const float qscale = 0.125f;  // 1/sqrt(64)
        gemm_bf16_kernel<<<grd, blk, 0, stream>>>(qin_bf, WqT, Qbuf, M, N, K, 1, qscale);
        gemm_bf16_kernel<<<grd, blk, 0, stream>>>(qin_bf, WkT, Kbuf, M, N, K, 1, 1.0f);
        gemm_bf16_kernel<<<grd, blk, 0, stream>>>(x_bf,   WvT, Vbuf, M, N, K, 2, 1.0f);
    }

    // 4) sliding-window attention, one wave per (b,h,16-query tile)
    {
        dim3 blk(32), grd(BATCH * NHEADS * (SEQ / 16));
        attn_window_kernel<<<grd, blk, 0, stream>>>(Qbuf, Kbuf, Vbuf, attn);
    }

    // 5) output projection -> f32
    {
        const int M = BATCH * SEQ, N = HID, K = HID;
        dim3 blk(128), grd(M / 64, N / 64);
        gemm_bf16_kernel<<<grd, blk, 0, stream>>>(attn, WoT, out, M, N, K, 0, 1.0f);
    }
}
